// LearnedBoardEncoder_11519102288245
// MI455X (gfx1250) — matrix operand order repaired and checked by reading, not compiled, not run on hardware
//
#include <hip/hip_runtime.h>
#include <hip/hip_bf16.h>

typedef __attribute__((ext_vector_type(16))) _Float16 v16h;
typedef __attribute__((ext_vector_type(8)))  float    v8f;
typedef __attribute__((ext_vector_type(4)))  float    v4f;

static constexpr int kD      = 256;
static constexpr int kPiece  = 7;
static constexpr int kColor  = 3;
static constexpr int kTok    = 67;   // 3 special tokens + 64 squares
static constexpr int kBatch  = 4096;
static constexpr float kEps  = 1e-5f;

// ---------------------------------------------------------------------------
// Kernel 1: P[c*7+p][n] = sum_k piece_table[p][k] * W[c][n][k] + b[c][n]
// 3 GEMMs of M=16(7 valid) x N=256 x K=256 via v_wmma_f32_16x16x32_f16.
// One wave (32 threads) per (color, 16-wide N tile): 3*16 = 48 blocks.
// All loads are unconditional float4s (clamped row index); invalid A rows are
// zeroed with a branch-free mask multiply so EXEC stays all-ones throughout.
// ---------------------------------------------------------------------------
__global__ __launch_bounds__(32) void precompute_proj_kernel(
    const float* __restrict__ piece_table,   // (7,256)
    const float* __restrict__ W,             // (3,256,256)  [c][out][in]
    const float* __restrict__ bias,          // (3,256)
    float* __restrict__ P)                   // (21,256) in workspace
{
    const int lane  = threadIdx.x;           // 0..31 (wave32)
    const int c     = blockIdx.x >> 4;       // color 0..2
    const int ntile = blockIdx.x & 15;       // 0..15
    const int m     = lane & 15;             // A-matrix row (piece id)
    const int hi    = lane >> 4;             // half-wave selector
    const int n     = ntile * 16 + m;        // output channel for this lane

    const bool  valid = (m < kPiece);
    const float zm    = valid ? 1.0f : 0.0f;           // branch-free row mask
    const float* __restrict__ arow = piece_table + (valid ? m : 0) * kD;
    const float* __restrict__ brow = W + (size_t)c * kD * kD + (size_t)n * kD;

    v8f acc = {};
#pragma unroll
    for (int step = 0; step < 8; ++step) {   // K = 8 * 32 = 256
        const int kb = step * 32;

        // A fragment (16x32 f16 ISA layout):
        //   elements 0-7  : K = kb + 8*hi + 0..7   (contiguous)
        //   elements 8-15 : K = kb + 16 + 8*hi + 0..7 (contiguous)
        const int a0 = kb + 8 * hi;
        const int a1 = kb + 16 + 8 * hi;
        v4f A0 = *(const v4f*)(arow + a0);
        v4f A1 = *(const v4f*)(arow + a0 + 4);
        v4f A2 = *(const v4f*)(arow + a1);
        v4f A3 = *(const v4f*)(arow + a1 + 4);
        A0 *= zm; A1 *= zm; A2 *= zm; A3 *= zm;

        // B fragment (32x16 f16, B[k][n] = W[c][n][k]):
        //   lane%16 = N; elements 0-7: K = kb + 8*hi + 0..7, elements 8-15: K = kb + 16 + 8*hi + 0..7
        const int b0 = kb + 8 * hi;
        const int b1 = kb + 16 + 8 * hi;
        v4f B0 = *(const v4f*)(brow + b0);
        v4f B1 = *(const v4f*)(brow + b0 + 4);
        v4f B2 = *(const v4f*)(brow + b1);
        v4f B3 = *(const v4f*)(brow + b1 + 4);

        v16h a, bm;
        a[0]  = (_Float16)A0.x; a[1]  = (_Float16)A0.y;
        a[2]  = (_Float16)A0.z; a[3]  = (_Float16)A0.w;
        a[4]  = (_Float16)A1.x; a[5]  = (_Float16)A1.y;
        a[6]  = (_Float16)A1.z; a[7]  = (_Float16)A1.w;
        a[8]  = (_Float16)A2.x; a[9]  = (_Float16)A2.y;
        a[10] = (_Float16)A2.z; a[11] = (_Float16)A2.w;
        a[12] = (_Float16)A3.x; a[13] = (_Float16)A3.y;
        a[14] = (_Float16)A3.z; a[15] = (_Float16)A3.w;

        bm[0]  = (_Float16)B0.x; bm[1]  = (_Float16)B0.y;
        bm[2]  = (_Float16)B0.z; bm[3]  = (_Float16)B0.w;
        bm[4]  = (_Float16)B1.x; bm[5]  = (_Float16)B1.y;
        bm[6]  = (_Float16)B1.z; bm[7]  = (_Float16)B1.w;
        bm[8]  = (_Float16)B2.x; bm[9]  = (_Float16)B2.y;
        bm[10] = (_Float16)B2.z; bm[11] = (_Float16)B2.w;
        bm[12] = (_Float16)B3.x; bm[13] = (_Float16)B3.y;
        bm[14] = (_Float16)B3.z; bm[15] = (_Float16)B3.w;

        // (neg_a, A, neg_b, B, c_mod, C, reuse_a, reuse_b)
        acc = __builtin_amdgcn_wmma_f32_16x16x32_f16(
            false, a, false, bm, (short)0, acc, false, false);
    }

    // D layout: VGPR r holds M = r (lanes 0-15) / M = r+8 (lanes 16-31), N = lane%16
    const float bb = bias[c * kD + n];
#pragma unroll
    for (int r = 0; r < 8; ++r) {
        const int row = r + 8 * hi;
        if (row < kPiece) {
            P[(c * kPiece + row) * kD + n] = acc[r] + bb;
        }
    }
}

// ---------------------------------------------------------------------------
// Kernel 2: fused token assembly + LayerNorm. One wave32 per token; each lane
// owns 8 channels (2 x float4). Wave reduction for mean/var; NT stores for the
// 281 MB streaming output (bigger than the 192 MB L2, never re-read).
// ---------------------------------------------------------------------------
__global__ __launch_bounds__(256) void encode_ln_kernel(
    const int*   __restrict__ piece_ids,      // (B,64)
    const int*   __restrict__ color_ids,      // (B,64)
    const int*   __restrict__ turn,           // (B,)
    const int*   __restrict__ castling,       // (B,)
    const int*   __restrict__ ep_file,        // (B,)
    const float* __restrict__ P,              // (21,256) precomputed
    const float* __restrict__ square_table,   // (64,256)
    const float* __restrict__ turn_table,     // (2,256)
    const float* __restrict__ castling_table, // (16,256)
    const float* __restrict__ ep_table,       // (9,256)
    const float* __restrict__ gamma,          // (256,)
    const float* __restrict__ beta,           // (256,)
    float*       __restrict__ out)            // (B,67,256)
{
    const int wid  = (int)((blockIdx.x * blockDim.x + threadIdx.x) >> 5);
    const int lane = threadIdx.x & 31;
    if (wid >= kBatch * kTok) return;

    const int b  = wid / kTok;
    const int t  = wid - b * kTok;
    const int d0 = lane * 8;

    v4f x0, x1;
    if (t >= 3) {
        const int s = t - 3;
        const int c = color_ids[b * 64 + s];   // wave-uniform broadcast loads
        const int p = piece_ids[b * 64 + s];
        const float* __restrict__ row = P + (c * kPiece + p) * kD;
        const float* __restrict__ sq  = square_table + s * kD;
        const v4f r0 = *(const v4f*)(row + d0);
        const v4f r1 = *(const v4f*)(row + d0 + 4);
        const v4f s0 = *(const v4f*)(sq + d0);
        const v4f s1 = *(const v4f*)(sq + d0 + 4);
        x0 = r0 + s0;
        x1 = r1 + s1;
    } else {
        const float* __restrict__ row;
        if (t == 0)      row = turn_table     + turn[b]     * kD;
        else if (t == 1) row = castling_table + castling[b] * kD;
        else             row = ep_table       + ep_file[b]  * kD;
        x0 = *(const v4f*)(row + d0);
        x1 = *(const v4f*)(row + d0 + 4);
    }

    // mean / variance over D=256 (8 per lane, then wave32 butterfly reduce)
    float s  = x0.x + x0.y + x0.z + x0.w + x1.x + x1.y + x1.z + x1.w;
    float ss = x0.x*x0.x + x0.y*x0.y + x0.z*x0.z + x0.w*x0.w
             + x1.x*x1.x + x1.y*x1.y + x1.z*x1.z + x1.w*x1.w;
#pragma unroll
    for (int off = 16; off > 0; off >>= 1) {
        s  += __shfl_xor(s,  off, 32);
        ss += __shfl_xor(ss, off, 32);
    }
    const float mean = s * (1.0f / kD);
    const float var  = ss * (1.0f / kD) - mean * mean;
    const float inv  = rsqrtf(var + kEps);

    const v4f g0  = *(const v4f*)(gamma + d0);
    const v4f g1  = *(const v4f*)(gamma + d0 + 4);
    const v4f be0 = *(const v4f*)(beta + d0);
    const v4f be1 = *(const v4f*)(beta + d0 + 4);

    v4f y0, y1;
    y0.x = (x0.x - mean) * inv * g0.x + be0.x;
    y0.y = (x0.y - mean) * inv * g0.y + be0.y;
    y0.z = (x0.z - mean) * inv * g0.z + be0.z;
    y0.w = (x0.w - mean) * inv * g0.w + be0.w;
    y1.x = (x1.x - mean) * inv * g1.x + be1.x;
    y1.y = (x1.y - mean) * inv * g1.y + be1.y;
    y1.z = (x1.z - mean) * inv * g1.z + be1.z;
    y1.w = (x1.w - mean) * inv * g1.w + be1.w;

    float* o = out + (size_t)wid * kD + d0;
    __builtin_nontemporal_store(y0, (v4f*)o);
    __builtin_nontemporal_store(y1, (v4f*)(o + 4));
}

extern "C" void kernel_launch(void* const* d_in, const int* in_sizes, int n_in,
                              void* d_out, int out_size, void* d_ws, size_t ws_size,
                              hipStream_t stream) {
    (void)in_sizes; (void)n_in; (void)out_size; (void)ws_size;

    const int*   piece_ids      = (const int*)d_in[0];
    const int*   color_ids      = (const int*)d_in[1];
    const int*   turn           = (const int*)d_in[2];
    const int*   castling       = (const int*)d_in[3];
    const int*   ep_file        = (const int*)d_in[4];
    const float* piece_table    = (const float*)d_in[5];
    const float* W              = (const float*)d_in[6];
    const float* bias           = (const float*)d_in[7];
    const float* square_table   = (const float*)d_in[8];
    const float* turn_table     = (const float*)d_in[9];
    const float* castling_table = (const float*)d_in[10];
    const float* ep_table       = (const float*)d_in[11];
    const float* gamma          = (const float*)d_in[12];
    const float* beta           = (const float*)d_in[13];
    float*       out            = (float*)d_out;
    float*       P              = (float*)d_ws;   // 21*256 floats = 21.5 KB

    // Stage 1: 21-row factorized projection via WMMA (3 colors x 16 N-tiles)
    precompute_proj_kernel<<<kColor * 16, 32, 0, stream>>>(piece_table, W, bias, P);

    // Stage 2: one wave per token; 8 waves / 256-thread block; exact division
    const int total_waves = kBatch * kTok;           // 274,432
    const int blocks      = total_waves / 8;         // 34,304 (exact)
    encode_ln_kernel<<<blocks, 256, 0, stream>>>(
        piece_ids, color_ids, turn, castling, ep_file,
        P, square_table, turn_table, castling_table, ep_table,
        gamma, beta, out);
}
